// MACAM_22849226015396
// MI455X (gfx1250) — matrix operand order repaired and checked by reading, not compiled, not run on hardware
//
#include <hip/hip_runtime.h>
#include <hip/hip_bf16.h>
#include <stdint.h>

// ---------------- Problem constants ----------------
#define Bb     16
#define CIN    512
#define HWSZ   4096
#define LL     64
#define WDIM   512
#define QDIM   512
#define EPSV   1e-5f

// ---------------- WMMA vector types ----------------
typedef __attribute__((ext_vector_type(16))) __bf16         v16bf;
typedef __attribute__((ext_vector_type(16))) unsigned short v16us;
typedef __attribute__((ext_vector_type(8)))  float          v8f;

union BF16Frag { v16us us; v16bf bf; };

__device__ __forceinline__ v8f wmma_bf16(v16bf a, v16bf b, v8f c) {
    return __builtin_amdgcn_wmma_f32_16x16x32_bf16(
        /*neg_a=*/false, a, /*neg_b=*/false, b,
        /*c_mod=*/(short)0, c, /*reuse_a=*/false, /*reuse_b=*/false);
}

// ---- CDNA5 async global->LDS copy (ASYNCcnt-tracked), inline asm ----
__device__ __forceinline__ void async_copy_b128(unsigned lds_byte_addr, const void* gaddr) {
    asm volatile("global_load_async_to_lds_b128 %0, %1, off"
                 :: "v"(lds_byte_addr), "v"((unsigned long long)gaddr)
                 : "memory");
}
__device__ __forceinline__ void wait_async0() {
    asm volatile("s_wait_asynccnt 0x0" ::: "memory");
}

// A fragment: 16x32 bf16 from f32 row-major A (lda floats/row).
// lane<16: m=lane,   K = k0 + {0..15}
// lane>=16: m=lane-16, K = k0 + {16..31}
__device__ __forceinline__ v16bf load_a_f32(const float* A, int lda, int mbase, int k0) {
    int lane = threadIdx.x & 31;
    int m    = mbase + (lane & 15);
    int kb   = (lane >> 4) * 16;
    const float* row = A + (size_t)m * lda + k0;
    float4 x0 = *(const float4*)(row + kb);
    float4 x1 = *(const float4*)(row + kb + 4);
    float4 x2 = *(const float4*)(row + kb + 8);
    float4 x3 = *(const float4*)(row + kb + 12);
    v16bf r;
    r[0]  = (__bf16)x0.x; r[1]  = (__bf16)x0.y; r[2]  = (__bf16)x0.z; r[3]  = (__bf16)x0.w;
    r[4]  = (__bf16)x1.x; r[5]  = (__bf16)x1.y; r[6]  = (__bf16)x1.z; r[7]  = (__bf16)x1.w;
    r[8]  = (__bf16)x2.x; r[9]  = (__bf16)x2.y; r[10] = (__bf16)x2.z; r[11] = (__bf16)x2.w;
    r[12] = (__bf16)x3.x; r[13] = (__bf16)x3.y; r[14] = (__bf16)x3.z; r[15] = (__bf16)x3.w;
    return r;
}

// B fragment 32x16 from f32 K-major global source (element [k][n], ldb floats/row)
__device__ __forceinline__ v16bf load_b_f32_kmajor(const float* Bm, int ldb, int k0, int nbase) {
    int lane = threadIdx.x & 31;
    int n    = nbase + (lane & 15);
    int kb   = (lane >> 4) * 16;
    const float* p = Bm + (size_t)(k0 + kb) * ldb + n;
    v16bf r;
#pragma unroll
    for (int i = 0; i < 16; ++i) r[i] = (__bf16)p[(size_t)i * ldb];
    return r;
}

// B fragment from N-major f32 source (element [n][k], ldn floats per n-row): contiguous K.
__device__ __forceinline__ v16bf load_b_f32_nmajor(const float* Bm, int ldn, int k0, int nbase) {
    int lane = threadIdx.x & 31;
    int n    = nbase + (lane & 15);
    int kb   = (lane >> 4) * 16;
    const float* row = Bm + (size_t)n * ldn + k0 + kb;
    float4 x0 = *(const float4*)(row);
    float4 x1 = *(const float4*)(row + 4);
    float4 x2 = *(const float4*)(row + 8);
    float4 x3 = *(const float4*)(row + 12);
    v16bf r;
    r[0]  = (__bf16)x0.x; r[1]  = (__bf16)x0.y; r[2]  = (__bf16)x0.z; r[3]  = (__bf16)x0.w;
    r[4]  = (__bf16)x1.x; r[5]  = (__bf16)x1.y; r[6]  = (__bf16)x1.z; r[7]  = (__bf16)x1.w;
    r[8]  = (__bf16)x2.x; r[9]  = (__bf16)x2.y; r[10] = (__bf16)x2.z; r[11] = (__bf16)x2.w;
    r[12] = (__bf16)x3.x; r[13] = (__bf16)x3.y; r[14] = (__bf16)x3.z; r[15] = (__bf16)x3.w;
    return r;
}

// B fragment from gathered rows: element [n][k] where row n -> wtgt[idx[n]]
__device__ __forceinline__ v16bf load_b_gather(const float* wtgt, const int* idxp, int k0, int nbase) {
    int lane = threadIdx.x & 31;
    int l    = nbase + (lane & 15);
    int kb   = (lane >> 4) * 16;
    const float* row = wtgt + (size_t)idxp[l] * WDIM + k0 + kb;
    float4 x0 = *(const float4*)(row);
    float4 x1 = *(const float4*)(row + 4);
    float4 x2 = *(const float4*)(row + 8);
    float4 x3 = *(const float4*)(row + 12);
    v16bf r;
    r[0]  = (__bf16)x0.x; r[1]  = (__bf16)x0.y; r[2]  = (__bf16)x0.z; r[3]  = (__bf16)x0.w;
    r[4]  = (__bf16)x1.x; r[5]  = (__bf16)x1.y; r[6]  = (__bf16)x1.z; r[7]  = (__bf16)x1.w;
    r[8]  = (__bf16)x2.x; r[9]  = (__bf16)x2.y; r[10] = (__bf16)x2.z; r[11] = (__bf16)x2.w;
    r[12] = (__bf16)x3.x; r[13] = (__bf16)x3.y; r[14] = (__bf16)x3.z; r[15] = (__bf16)x3.w;
    return r;
}

// B fragment from LDS f32 tile (pitch 128 floats), element [k][n]
__device__ __forceinline__ v16bf load_b_lds_f32(const float* S, int nloc) {
    int lane = threadIdx.x & 31;
    int n    = nloc + (lane & 15);
    int kb   = (lane >> 4) * 16;
    const float* p = S + (size_t)kb * 128 + n;
    v16bf r;
#pragma unroll
    for (int i = 0; i < 16; ++i) r[i] = (__bf16)p[(size_t)i * 128];
    return r;
}

// B fragment from LDS bf16 tile (pitch 128 elems), element [k][n]
__device__ __forceinline__ v16bf load_b_lds_bf16(const unsigned short* S, int k0, int nloc) {
    int lane = threadIdx.x & 31;
    int n    = nloc + (lane & 15);
    int kb   = (lane >> 4) * 16;
    const unsigned short* p = S + (size_t)(k0 + kb) * 128 + n;
    BF16Frag r;
#pragma unroll
    for (int i = 0; i < 16; ++i) r.us[i] = p[(size_t)i * 128];
    return r.bf;
}

// ============================================================
// K1: per-(b,c) mean / rsqrt(var+eps) over HW
// ============================================================
__global__ void meanvar_kernel(const float* __restrict__ h,
                               float* __restrict__ mu, float* __restrict__ rsig) {
    int bc  = blockIdx.x;
    int tid = threadIdx.x;
    const float* row = h + (size_t)bc * HWSZ;
    float s = 0.f, s2 = 0.f;
    for (int i = tid; i < HWSZ; i += 256) { float v = row[i]; s += v; s2 += v * v; }
    __shared__ float ss[256], ss2[256];
    ss[tid] = s; ss2[tid] = s2;
    __syncthreads();
    for (int st = 128; st > 0; st >>= 1) {
        if (tid < st) { ss[tid] += ss[tid + st]; ss2[tid] += ss2[tid + st]; }
        __syncthreads();
    }
    if (tid == 0) {
        float m   = ss[0]  * (1.f / HWSZ);
        float var = ss2[0] * (1.f / HWSZ) - m * m;
        mu[bc]   = m;
        rsig[bc] = rsqrtf(var + EPSV);
    }
}

// ============================================================
// K2: k[b] (64x512) = w_source[b] @ fc_k_w^T + fc_k_b
// ============================================================
__global__ void k_gemm_kernel(const float* __restrict__ wsrc,
                              const float* __restrict__ fckw,
                              const float* __restrict__ fckb,
                              float* __restrict__ k_out) {
    int b = blockIdx.x;
    int wv = threadIdx.x >> 5, lane = threadIdx.x & 31;
    int mtile  = (wv & 3) * 16;
    int nbase0 = blockIdx.y * 128 + (wv >> 2) * 64;
    const float* A = wsrc + (size_t)b * LL * WDIM;
    v8f acc[4] = {};
    for (int k0 = 0; k0 < WDIM; k0 += 32) {
        v16bf a = load_a_f32(A, WDIM, mtile, k0);
#pragma unroll
        for (int j = 0; j < 4; ++j) {
            v16bf bb = load_b_f32_nmajor(fckw, WDIM, k0, nbase0 + j * 16);
            acc[j] = wmma_bf16(a, bb, acc[j]);
        }
    }
    int hi = lane >> 4, nl = lane & 15;
    float* O = k_out + (size_t)b * LL * QDIM;
#pragma unroll
    for (int j = 0; j < 4; ++j) {
        int n = nbase0 + j * 16 + nl;
        float bias = fckb[n];
#pragma unroll
        for (int r = 0; r < 8; ++r) {
            int m = mtile + hi * 8 + r;
            O[(size_t)m * QDIM + n] = acc[j][r] + bias;
        }
    }
}

// ============================================================
// K3: kk[b] (64x512) = k[b] @ conv_w ([o][c] K-major)
// ============================================================
__global__ void kk_gemm_kernel(const float* __restrict__ k_in,
                               const float* __restrict__ convw,
                               float* __restrict__ kk_out) {
    int b = blockIdx.x;
    int wv = threadIdx.x >> 5, lane = threadIdx.x & 31;
    int mtile  = (wv & 3) * 16;
    int nbase0 = blockIdx.y * 128 + (wv >> 2) * 64;
    const float* A = k_in + (size_t)b * LL * QDIM;
    v8f acc[4] = {};
    for (int k0 = 0; k0 < QDIM; k0 += 32) {
        v16bf a = load_a_f32(A, QDIM, mtile, k0);
#pragma unroll
        for (int j = 0; j < 4; ++j) {
            v16bf bb = load_b_f32_kmajor(convw, CIN, k0, nbase0 + j * 16);
            acc[j] = wmma_bf16(a, bb, acc[j]);
        }
    }
    int hi = lane >> 4, nl = lane & 15;
    float* O = kk_out + (size_t)b * LL * CIN;
#pragma unroll
    for (int j = 0; j < 4; ++j) {
        int n = nbase0 + j * 16 + nl;
#pragma unroll
        for (int r = 0; r < 8; ++r) {
            int m = mtile + hi * 8 + r;
            O[(size_t)m * CIN + n] = acc[j][r];
        }
    }
}

// K3b: kb[b][l] = k[b][l] . conv_b
__global__ void kb_kernel(const float* __restrict__ k_in,
                          const float* __restrict__ convb,
                          float* __restrict__ kb) {
    int b = blockIdx.x, l = threadIdx.x;
    const float* row = k_in + ((size_t)b * LL + l) * QDIM;
    float s = 0.f;
    for (int o = 0; o < QDIM; ++o) s += row[o] * convb[o];
    kb[b * LL + l] = s;
}

// ============================================================
// K4: scores -> left_scores argmax -> indices  (one block per batch)
// ============================================================
__global__ void score_argmax_kernel(const float* __restrict__ wsrc,
                                    const float* __restrict__ wtgt,
                                    int* __restrict__ idx_out) {
    __shared__ float sc[LL][LL + 1];
    __shared__ float colsum[LL];
    int b = blockIdx.x, tid = threadIdx.x;
    const float* S = wsrc + (size_t)b * LL * WDIM;
    const float* T = wtgt + (size_t)b * LL * WDIM;
    for (int e = tid; e < LL * LL; e += 256) {
        int l = e >> 6, m = e & 63;
        const float* sr = S + (size_t)l * WDIM;
        const float* tr = T + (size_t)m * WDIM;
        float s = 0.f;
        for (int d = 0; d < WDIM; ++d) s += sr[d] * tr[d];
        sc[l][m] = s;
    }
    __syncthreads();
    if (tid < LL) {
        float s = 0.f;
        for (int l = 0; l < LL; ++l) s += sc[l][tid];
        colsum[tid] = s;
    }
    __syncthreads();
    if (tid < LL) {
        int l = tid;
        float best = colsum[0] - sc[l][0];
        int bi = 0;
        for (int m = 1; m < LL; ++m) {
            float v = colsum[m] - sc[l][m];
            if (v > best) { best = v; bi = m; }   // first-max, like jnp.argmax
        }
        idx_out[b * LL + l] = bi;
    }
}

// ============================================================
// K5: bgT[b] (1024x64) = fc_w @ w_alloc^T + fc_b (stored transposed)
// ============================================================
__global__ void bg_gemm_kernel(const float* __restrict__ fcw,
                               const float* __restrict__ fcb,
                               const float* __restrict__ wtgt,
                               const int* __restrict__ idx,
                               float* __restrict__ bgT) {
    int b = blockIdx.x;
    int wv = threadIdx.x >> 5, lane = threadIdx.x & 31;
    int mbase = blockIdx.y * 128 + wv * 16;
    const float* T = wtgt + (size_t)b * LL * WDIM;
    const int* ip  = idx + b * LL;
    v8f acc[4] = {};
    for (int k0 = 0; k0 < WDIM; k0 += 32) {
        v16bf a = load_a_f32(fcw, WDIM, mbase, k0);
#pragma unroll
        for (int j = 0; j < 4; ++j) {
            v16bf bb = load_b_gather(T, ip, k0, j * 16);
            acc[j] = wmma_bf16(a, bb, acc[j]);
        }
    }
    int hi = lane >> 4, nl = lane & 15;
    float* O = bgT + (size_t)b * (2 * CIN) * LL;
#pragma unroll
    for (int r = 0; r < 8; ++r) {
        int c2 = mbase + hi * 8 + r;
        float bias = fcb[c2];
#pragma unroll
        for (int j = 0; j < 4; ++j) {
            int l = j * 16 + nl;
            O[(size_t)c2 * LL + l] = acc[j][r] + bias;
        }
    }
}

// ============================================================
// K6: attn[b] (64x4096, bf16) = kk[b] (64x512) @ h[b] (512x4096) + kb bcast
// h streamed through LDS via double-buffered async global->LDS B128 copies.
// grid (B, 32): 128 cols per block, 256 threads (8 waves)
// ============================================================
__global__ void attn_gemm_kernel(const float* __restrict__ kk,
                                 const float* __restrict__ h,
                                 const float* __restrict__ kb,
                                 __bf16* __restrict__ attn) {
    __shared__ alignas(16) float sh[2][32 * 128];   // 2 x 16KB K-slabs of h
    int b   = blockIdx.x;
    int tid = threadIdx.x;
    int wv  = tid >> 5, lane = tid & 31;
    int mtile = (wv & 3) * 16;
    int nloc0 = (wv >> 2) * 64;                      // local col base (0 or 64)
    int pbase = blockIdx.y * 128;
    const float* A = kk + (size_t)b * LL * CIN;
    const float* H = h + (size_t)b * CIN * HWSZ + pbase;

    // issue slab 0: 32 rows x 128 floats = 1024 x 16B chunks, 4 per thread
    {
        unsigned base = (unsigned)(size_t)&sh[0][0];
#pragma unroll
        for (int i = 0; i < 4; ++i) {
            int c = tid + 256 * i;
            int row = c >> 5, col = (c & 31) * 4;
            async_copy_b128(base + (unsigned)(row * 128 + col) * 4,
                            H + (size_t)row * HWSZ + col);
        }
    }

    v8f acc[4] = {};
    for (int ks = 0; ks < 16; ++ks) {
        int par = ks & 1;
        wait_async0();          // this wave's copies have landed in LDS
        __syncthreads();        // all waves' copies visible
        if (ks + 1 < 16) {      // issue next slab into the other buffer
            unsigned base = (unsigned)(size_t)&sh[par ^ 1][0];
            const float* Hn = H + (size_t)(ks + 1) * 32 * HWSZ;
#pragma unroll
            for (int i = 0; i < 4; ++i) {
                int c = tid + 256 * i;
                int row = c >> 5, col = (c & 31) * 4;
                async_copy_b128(base + (unsigned)(row * 128 + col) * 4,
                                Hn + (size_t)row * HWSZ + col);
            }
        }
        v16bf a = load_a_f32(A, CIN, mtile, ks * 32);
        const float* S = &sh[par][0];
#pragma unroll
        for (int j = 0; j < 4; ++j) {
            v16bf bb = load_b_lds_f32(S, nloc0 + j * 16);
            acc[j] = wmma_bf16(a, bb, acc[j]);
        }
        __syncthreads();        // reads done before buffer reuse
    }

    int hi = lane >> 4, nl = lane & 15;
    const float* kbp = kb + b * LL;
    __bf16* O = attn + (size_t)b * LL * HWSZ;
#pragma unroll
    for (int r = 0; r < 8; ++r) {
        int l = mtile + hi * 8 + r;
        float bias = kbp[l];
#pragma unroll
        for (int j = 0; j < 4; ++j) {
            int p = pbase + nloc0 + j * 16 + nl;
            O[(size_t)l * HWSZ + p] = (__bf16)(acc[j][r] + bias);
        }
    }
}

// ============================================================
// K7: fused modulation-map GEMM + InstanceNorm + AdaIN
// attn tile (64x128 bf16 = 16KB) staged into LDS with async copies once.
// grid (B, 8, 32): 64 channels x 128 positions per block
// ============================================================
__global__ void out_kernel(const float* __restrict__ bgT,
                           const unsigned short* __restrict__ attn,
                           const float* __restrict__ h,
                           const float* __restrict__ mu,
                           const float* __restrict__ rsig,
                           const float* __restrict__ inw,
                           const float* __restrict__ inb,
                           float* __restrict__ out) {
    __shared__ alignas(16) unsigned short s_at[64 * 128];   // attn tile, bf16 bits
    int b   = blockIdx.x;
    int tid = threadIdx.x;
    int wv  = tid >> 5, lane = tid & 31;
    int cbase = blockIdx.y * 64 + (wv & 3) * 16;
    int ploc0 = (wv >> 2) * 64;
    int pbase = blockIdx.z * 128;
    const float* BG = bgT + (size_t)b * (2 * CIN) * LL;
    const unsigned short* ATg = attn + (size_t)b * LL * HWSZ + pbase;

    // stage attn tile: 64 rows x 256B = 1024 x 16B chunks, 4 per thread
    {
        unsigned base = (unsigned)(size_t)&s_at[0];
#pragma unroll
        for (int i = 0; i < 4; ++i) {
            int c = tid + 256 * i;
            int row = c >> 4, col = (c & 15) * 8;           // 8 ushorts = 16B
            async_copy_b128(base + (unsigned)(row * 128 + col) * 2,
                            ATg + (size_t)row * HWSZ + col);
        }
    }
    wait_async0();
    __syncthreads();

    v8f accB[4] = {}, accG[4] = {};
#pragma unroll
    for (int k0 = 0; k0 < LL; k0 += 32) {
        v16bf ab = load_a_f32(BG, LL, cbase, k0);                    // beta rows
        v16bf ag = load_a_f32(BG + (size_t)CIN * LL, LL, cbase, k0); // gamma rows
#pragma unroll
        for (int j = 0; j < 4; ++j) {
            v16bf bb = load_b_lds_bf16(s_at, k0, ploc0 + j * 16);
            accB[j] = wmma_bf16(ab, bb, accB[j]);
            accG[j] = wmma_bf16(ag, bb, accG[j]);
        }
    }

    int hi = lane >> 4, nl = lane & 15;
    const float* H  = h + (size_t)b * CIN * HWSZ;
    float*       O  = out + (size_t)b * CIN * HWSZ;
    const float* mp = mu + b * CIN;
    const float* rp = rsig + b * CIN;
#pragma unroll
    for (int r = 0; r < 8; ++r) {
        int c = cbase + hi * 8 + r;
        float m = mp[c], rs = rp[c], gw = inw[c], gb = inb[c];
#pragma unroll
        for (int j = 0; j < 4; ++j) {
            int p = pbase + ploc0 + j * 16 + nl;
            float hv = H[(size_t)c * HWSZ + p];
            float hn = (hv - m) * rs * gw + gb;
            O[(size_t)c * HWSZ + p] = hn * accG[j][r] + accB[j][r];
        }
    }
}

// ============================================================
extern "C" void kernel_launch(void* const* d_in, const int* in_sizes, int n_in,
                              void* d_out, int out_size, void* d_ws, size_t ws_size,
                              hipStream_t stream) {
    (void)in_sizes; (void)n_in; (void)out_size; (void)ws_size;
    const float* h      = (const float*)d_in[0];
    const float* wsrc   = (const float*)d_in[1];
    const float* wtgt   = (const float*)d_in[2];
    const float* convw  = (const float*)d_in[3];   // [QDIM, CIN]
    const float* convb  = (const float*)d_in[4];
    const float* fckw   = (const float*)d_in[5];   // [QDIM, WDIM]
    const float* fckb   = (const float*)d_in[6];
    const float* fcw    = (const float*)d_in[7];   // [2*CIN, WDIM]
    const float* fcb    = (const float*)d_in[8];
    const float* in_w   = (const float*)d_in[9];
    const float* in_b   = (const float*)d_in[10];
    float* out = (float*)d_out;

    // -------- workspace layout (floats) --------
    float* ws = (float*)d_ws;
    size_t off = 0;
    float* ws_k    = ws + off; off += (size_t)Bb * LL * QDIM;        // 524288
    float* ws_kk   = ws + off; off += (size_t)Bb * LL * CIN;         // 524288
    float* ws_kb   = ws + off; off += (size_t)Bb * LL;               // 1024
    float* ws_bgT  = ws + off; off += (size_t)Bb * 2 * CIN * LL;     // 1048576
    float* ws_mu   = ws + off; off += (size_t)Bb * CIN;              // 8192
    float* ws_rsig = ws + off; off += (size_t)Bb * CIN;              // 8192
    int*   ws_idx  = (int*)(ws + off); off += (size_t)Bb * LL;       // 1024
    __bf16* ws_attn = (__bf16*)(ws + off);                           // 16*64*4096 bf16
    // total ~16.1 MB

    meanvar_kernel<<<dim3(Bb * CIN), 256, 0, stream>>>(h, ws_mu, ws_rsig);

    k_gemm_kernel<<<dim3(Bb, QDIM / 128), 256, 0, stream>>>(wsrc, fckw, fckb, ws_k);

    kb_kernel<<<dim3(Bb), LL, 0, stream>>>(ws_k, convb, ws_kb);

    kk_gemm_kernel<<<dim3(Bb, CIN / 128), 256, 0, stream>>>(ws_k, convw, ws_kk);

    score_argmax_kernel<<<dim3(Bb), 256, 0, stream>>>(wsrc, wtgt, ws_idx);

    bg_gemm_kernel<<<dim3(Bb, (2 * CIN) / 128), 256, 0, stream>>>(fcw, fcb, wtgt, ws_idx, ws_bgT);

    attn_gemm_kernel<<<dim3(Bb, HWSZ / 128), 256, 0, stream>>>(ws_kk, h, ws_kb, ws_attn);

    out_kernel<<<dim3(Bb, CIN / 64, HWSZ / 128), 256, 0, stream>>>(
        ws_bgT, (const unsigned short*)ws_attn, h, ws_mu, ws_rsig, in_w, in_b, out);
}
